// Self_Attention_4415226380651
// MI455X (gfx1250) — matrix operand order repaired and checked by reading, not compiled, not run on hardware
//
#include <hip/hip_runtime.h>
#include <hip/hip_bf16.h>

typedef __bf16 bf16_t;
typedef __attribute__((ext_vector_type(16))) __bf16 v16bf;
typedef __attribute__((ext_vector_type(8)))  __bf16 v8bf;
typedef __attribute__((ext_vector_type(8)))  float  v8f;
typedef __attribute__((ext_vector_type(4)))  unsigned v4u;
typedef __attribute__((ext_vector_type(8)))  int      v8i;
typedef __attribute__((ext_vector_type(4)))  int      v4i;

#define BATCH  16
#define CCH    256
#define CQK    32
#define NSP    2304      // 48*48
#define JTILE  32
#define NWAVES 8
#define RBLK   (NSP/16)  // 144 row blocks per batch
#define NT     (NSP/JTILE)  // 72 j-tiles

static_assert(RBLK % NWAVES == 0, "all waves of a block share one batch");

// Tensor Data Mover availability (this toolchain: 6-arg builtin,
// (v4u g0, v8i g1, v4i g2, v4i g3, v8i extra, i32 cpol))
#if defined(__has_builtin)
#  if __has_builtin(__builtin_amdgcn_tensor_load_to_lds)
#    define HAS_TDM 1
#  endif
#endif
#ifndef HAS_TDM
#  define HAS_TDM 0
#endif

// ---------------------------------------------------------------------------
// WMMA fragment helpers (contiguous k = 16*half + s in both A and B, wave32)
// ---------------------------------------------------------------------------

// A fragment, 16x32 bf16, source row-major [16][row_stride].
// lane<16 : row=lane,    k in {0..15}
// lane>=16: row=lane-16, k in {16..31}
static __device__ __forceinline__ v16bf wmma_a_rowmajor(const bf16_t* base,
                                                        int row_stride, int lane) {
  const int m   = lane & 15;
  const int off = (lane < 16) ? 0 : 16;
  v8bf lo = *(const v8bf*)(base + m * row_stride + off);
  v8bf hi = *(const v8bf*)(base + m * row_stride + 8 + off);
  return __builtin_shufflevector(lo, hi, 0, 1, 2, 3, 4, 5, 6, 7,
                                 8, 9, 10, 11, 12, 13, 14, 15);
}

// B fragment, 32x16 bf16, each column n has its 32 k-values contiguous at
// (base + n*col_stride).  lane<16: col=lane, k=0..15; lane>=16: k=16..31.
static __device__ __forceinline__ v16bf wmma_b_kcontig(const bf16_t* base,
                                                       size_t col_stride, int lane) {
  const int n    = lane & 15;
  const int koff = (lane < 16) ? 0 : 16;
  return *(const v16bf*)(base + (size_t)n * col_stride + koff);
}

static __device__ __forceinline__ v8f wmma_bf16(v16bf a, v16bf b, v8f c) {
  return __builtin_amdgcn_wmma_f32_16x16x32_bf16(false, a, false, b,
                                                 (short)0, c, false, false);
}

// Generic pointers into LDS carry the LDS byte offset in addr[31:0]
// (AS3->flat addrspacecast; ISA 10.2 aperture rules).
static __device__ __forceinline__ unsigned lds_off(const void* p) {
  return (unsigned)(unsigned long long)p;
}

// Issue a 2D TDM load: tile (tile_d0 x tile_d1) elements from a row-major
// tensor with row stride `stride0` (elements), element size 2^ds_code bytes.
static __device__ __forceinline__ void tdm_load_2d(unsigned lds_byte_off,
                                                   unsigned long long gaddr,
                                                   unsigned tensor_d0,
                                                   unsigned tensor_d1,
                                                   unsigned tile_d0,
                                                   unsigned tile_d1,
                                                   unsigned stride0,
                                                   unsigned ds_code) {
#if HAS_TDM
  // D# group0: count=1 | lds_addr | global_addr[56:0] | type=2
  v4u g0 = {1u, lds_byte_off, (unsigned)gaddr,
            (unsigned)((gaddr >> 32) & 0x1FFFFFFull) | (2u << 30)};
  // D# group1 bit-packing per ISA 8.4
  v8i g1 = {(int)(ds_code << 16),
            (int)(tensor_d0 << 16),          // tensor_dim0[15:0] @ [63:48]
            (int)((tensor_d0 >> 16) | (tensor_d1 << 16)),
            (int)((tensor_d1 >> 16) | (tile_d0 << 16)),
            (int)(tile_d1),                  // tile_dim1 @ [143:128]
            (int)(stride0),                  // tensor_dim0_stride[31:0]
            0, 0};
  v4i gz4 = {0, 0, 0, 0};
  v8i gz8 = {0, 0, 0, 0, 0, 0, 0, 0};
  __builtin_amdgcn_tensor_load_to_lds(g0, g1, gz4, gz4, gz8, 0);
#else
  (void)lds_byte_off; (void)gaddr; (void)tensor_d0; (void)tensor_d1;
  (void)tile_d0; (void)tile_d1; (void)stride0; (void)ds_code;
#endif
}

// ---------------------------------------------------------------------------
// Kernel 1: fp32 -> bf16 weight conversion (Wv only)
// ---------------------------------------------------------------------------
__global__ void cvt_kernel(const float* __restrict__ src, bf16_t* __restrict__ dst,
                           int n) {
  int i = blockIdx.x * 256 + threadIdx.x;
  if (i < n) dst[i] = (bf16_t)src[i];
}

// ---------------------------------------------------------------------------
// Kernel 2: Q/K projection (small: 1.2 GFLOP) -> Qbf[b][n][32], Kbf[b][n][32]
// ---------------------------------------------------------------------------
__global__ __launch_bounds__(256) void qkproj_kernel(
    const float* __restrict__ x, const float* __restrict__ Wq,
    const float* __restrict__ bq, const float* __restrict__ Wk,
    const float* __restrict__ bk, bf16_t* __restrict__ Qbf,
    bf16_t* __restrict__ Kbf) {
  int gid = blockIdx.x * 256 + threadIdx.x;  // 0 .. B*NSP-1
  int b = gid / NSP;
  int n = gid % NSP;
  float aq[CQK], ak[CQK];
#pragma unroll
  for (int o = 0; o < CQK; o++) { aq[o] = bq[o]; ak[o] = bk[o]; }
  for (int c = 0; c < CCH; c++) {
    float xv = x[((size_t)b * CCH + c) * NSP + n];
#pragma unroll
    for (int o = 0; o < CQK; o++) {
      aq[o] = fmaf(Wq[o * CCH + c], xv, aq[o]);
      ak[o] = fmaf(Wk[o * CCH + c], xv, ak[o]);
    }
  }
  size_t base = ((size_t)b * NSP + n) * CQK;
#pragma unroll
  for (int o = 0; o < CQK; o++) {
    Qbf[base + o] = (bf16_t)aq[o];
    Kbf[base + o] = (bf16_t)ak[o];
  }
}

// ---------------------------------------------------------------------------
// Kernel 3: V projection via WMMA -> Vbf[b][o][n] (channel-major, bf16)
// Block-cooperative: one block per (batch, 16-column n-tile).  The x tile
// [256c x 16n] fp32 is staged once via TDM, transposed/converted to bf16 in
// LDS, then each wave computes 2 o-tiles with all-LDS B fragments.
// ---------------------------------------------------------------------------
__global__ __launch_bounds__(256) void vproj_kernel(
    const float* __restrict__ x, const bf16_t* __restrict__ Wv_bf,
    const float* __restrict__ bv, bf16_t* __restrict__ Vbf) {
  __shared__ __attribute__((aligned(64))) float  Xt[CCH][16];  // 16KB TDM dest
  __shared__ __attribute__((aligned(64))) bf16_t Xb[16][CCH];  // 8KB  k-contig

  const int tid  = threadIdx.x;
  const int lane = tid & 31;
  const int wave = tid >> 5;
  const int b  = blockIdx.x / (NSP / 16);
  const int n0 = (blockIdx.x % (NSP / 16)) * 16;
  const int hi = (lane >= 16) ? 1 : 0;

  const float* xtile = x + ((size_t)b * CCH) * NSP + n0;

  // stage x tile: rows c=0..255, 16 contiguous fp32 each
#if HAS_TDM
  if (wave == 0) {
    tdm_load_2d(lds_off(&Xt[0][0]), (unsigned long long)xtile,
                NSP, CCH, 16, CCH, NSP, /*4B*/ 2);
    __builtin_amdgcn_s_wait_tensorcnt(0);
  }
#else
  {
    unsigned long long ga =
        (unsigned long long)xtile + (unsigned long long)tid * (NSP * 4);
    unsigned lo = lds_off(&Xt[0][0]) + (unsigned)tid * 64;
#pragma unroll
    for (int i = 0; i < 4; i++)
      asm volatile("global_load_async_to_lds_b128 %0, %1, off"
                   :: "v"(lo + i * 16), "v"(ga + i * 16) : "memory");
    asm volatile("s_wait_asynccnt 0x0" ::: "memory");
  }
#endif
  __syncthreads();

  // transpose + convert: Xb[n][c] = bf16(Xt[c][n]);  thread owns column c=tid
#pragma unroll
  for (int n = 0; n < 16; n++) Xb[n][tid] = (bf16_t)Xt[tid][n];
  __syncthreads();

  const int nloc = lane & 15;
  const int n = n0 + nloc;
#pragma unroll
  for (int half = 0; half < 2; half++) {
    const int o0 = (wave * 2 + half) * 16;
    v8f acc = {0.f, 0.f, 0.f, 0.f, 0.f, 0.f, 0.f, 0.f};
    for (int c0 = 0; c0 < CCH; c0 += 32) {
      v16bf afr = wmma_a_rowmajor(Wv_bf + (size_t)o0 * CCH + c0, CCH, lane);
      v16bf bfr = wmma_b_kcontig(&Xb[0][c0], CCH, lane);
      acc = wmma_bf16(afr, bfr, acc);
    }
#pragma unroll
    for (int r = 0; r < 8; r++) {
      int o = o0 + r + hi * 8;
      float v = acc[r] + bv[o];
      Vbf[((size_t)b * CCH + o) * NSP + n] = (bf16_t)v;
    }
  }
}

// ---------------------------------------------------------------------------
// Kernel 4: fused flash attention.  8 waves/block, one 16-row query block per
// wave; K/V j-tiles staged in LDS once per block via CDNA5 async paths:
//   K tile (2KB):  global_load_async_to_lds_b64, 1 per thread  (ASYNCcnt)
//   V tile (16KB): one TDM descriptor issued by wave 0         (TENSORcnt)
// double-buffered, partial-count waits, online softmax, O in 16 f32 C-tiles.
// ---------------------------------------------------------------------------
__global__ __launch_bounds__(256) void attn_kernel(
    const bf16_t* __restrict__ Qbf, const bf16_t* __restrict__ Kbf,
    const bf16_t* __restrict__ Vbf, const float* __restrict__ inF,
    const float* __restrict__ gamma, float* __restrict__ out) {
  __shared__ __attribute__((aligned(64))) bf16_t Ktile[2][JTILE][CQK]; // 2x2KB
  __shared__ __attribute__((aligned(64))) bf16_t Vtile[2][CCH][JTILE]; // 2x16KB
  __shared__ __attribute__((aligned(64))) bf16_t Pbuf[NWAVES][16][JTILE];

  const int tid  = threadIdx.x;
  const int lane = tid & 31;
  const int wave = tid >> 5;
  const int rb = blockIdx.x * NWAVES + wave;  // 0 .. B*RBLK-1
  const int b = rb / RBLK;                    // uniform across the block
  const int r0 = (rb % RBLK) * 16;
  const int hi = (lane >= 16) ? 1 : 0;
  const int col = lane & 15;

  const bf16_t* kbase_b = Kbf + (size_t)b * NSP * CQK;
  const bf16_t* vbase_b = Vbf + (size_t)b * CCH * NSP;

  const unsigned koff0 = lds_off(&Ktile[0][0][0]);
  const unsigned koff1 = lds_off(&Ktile[1][0][0]);
  const unsigned voff0 = lds_off(&Vtile[0][0][0]);
  const unsigned voff1 = lds_off(&Vtile[1][0][0]);

  // ---- issue one j-tile's staging (K async per-thread, V via TDM) ----
  auto issue_tile = [&](int jt, int p) {
    // K tile: contiguous 2KB (Kbf is [n][32] row-major) -> one b64 per thread
    {
      unsigned long long ga = (unsigned long long)(kbase_b) +
                              (unsigned long long)jt * (JTILE * CQK * 2) +
                              (unsigned long long)tid * 8;
      unsigned lo = (p ? koff1 : koff0) + (unsigned)tid * 8;
      asm volatile("global_load_async_to_lds_b64 %0, %1, off"
                   :: "v"(lo), "v"(ga) : "memory");
    }
#if HAS_TDM
    if (wave == 0) {
      unsigned long long ga = (unsigned long long)(vbase_b) +
                              (unsigned long long)jt * (JTILE * 2);
      tdm_load_2d(p ? voff1 : voff0, ga, NSP, CCH, JTILE, CCH, NSP, /*2B*/ 1);
    }
#else
    // fallback: V tile rows via async b128 (4 per thread, c = tid)
    {
      unsigned long long ga = (unsigned long long)(vbase_b) +
          ((unsigned long long)tid * NSP + (unsigned long long)jt * JTILE) * 2;
      unsigned lo = (p ? voff1 : voff0) + (unsigned)tid * (JTILE * 2);
#pragma unroll
      for (int i = 0; i < 4; i++)
        asm volatile("global_load_async_to_lds_b128 %0, %1, off"
                     :: "v"(lo + i * 16), "v"(ga + i * 16) : "memory");
    }
#endif
  };

  // Q A-fragment, loaded once (16 rows x K=32)
  v16bf qa = wmma_a_rowmajor(Qbf + ((size_t)b * NSP + r0) * CQK, CQK, lane);

  v8f acc[CCH / 16];
#pragma unroll
  for (int t = 0; t < CCH / 16; t++)
    acc[t] = (v8f){0.f, 0.f, 0.f, 0.f, 0.f, 0.f, 0.f, 0.f};

  float mrow[8], lrow[8];
#pragma unroll
  for (int r = 0; r < 8; r++) { mrow[r] = -1e30f; lrow[r] = 0.f; }

  // prologue: stage tiles 0 and 1
  issue_tile(0, 0);
  issue_tile(1, 1);

  for (int jt = 0; jt < NT; ++jt) {
    const int p = jt & 1;

    // drain this tile (in-order completion => partial-count wait is exact)
#if HAS_TDM
    if (jt + 1 < NT) asm volatile("s_wait_asynccnt 0x1" ::: "memory");
    else             asm volatile("s_wait_asynccnt 0x0" ::: "memory");
    if (wave == 0) {
      if (jt + 1 < NT) __builtin_amdgcn_s_wait_tensorcnt(1);
      else             __builtin_amdgcn_s_wait_tensorcnt(0);
    }
#else
    if (jt + 1 < NT) asm volatile("s_wait_asynccnt 0x5" ::: "memory");
    else             asm volatile("s_wait_asynccnt 0x0" ::: "memory");
#endif
    __syncthreads();

    // ---- S = Q * K^T for 32 key columns (two 16-wide tiles) ----
    const bf16_t* kt = &Ktile[p][0][0];
    v16bf kb0 = wmma_b_kcontig(kt, CQK, lane);
    v16bf kb1 = wmma_b_kcontig(kt + 16 * CQK, CQK, lane);
    v8f z = {0.f, 0.f, 0.f, 0.f, 0.f, 0.f, 0.f, 0.f};
    v8f s0 = wmma_bf16(qa, kb0, z);
    v8f s1 = wmma_bf16(qa, kb1, z);

    // ---- online softmax (16-lane half reduce; half owns rows hi*8+r) ----
    float mnew[8], scale[8];
#pragma unroll
    for (int r = 0; r < 8; r++) {
      float t = fmaxf(s0[r], s1[r]);
      t = fmaxf(t, __shfl_xor(t, 1, 32));
      t = fmaxf(t, __shfl_xor(t, 2, 32));
      t = fmaxf(t, __shfl_xor(t, 4, 32));
      t = fmaxf(t, __shfl_xor(t, 8, 32));
      mnew[r] = fmaxf(mrow[r], t);
      scale[r] = __expf(mrow[r] - mnew[r]);
    }
    float p0[8], p1[8];
#pragma unroll
    for (int r = 0; r < 8; r++) {
      p0[r] = __expf(s0[r] - mnew[r]);
      p1[r] = __expf(s1[r] - mnew[r]);
      float s = p0[r] + p1[r];
      s += __shfl_xor(s, 1, 32);
      s += __shfl_xor(s, 2, 32);
      s += __shfl_xor(s, 4, 32);
      s += __shfl_xor(s, 8, 32);
      lrow[r] = lrow[r] * scale[r] + s;
      mrow[r] = mnew[r];
    }

    // ---- rescale O accumulators ----
#pragma unroll
    for (int t = 0; t < CCH / 16; t++)
#pragma unroll
      for (int r = 0; r < 8; r++) acc[t][r] = acc[t][r] * scale[r];

    // ---- C-layout P -> row-major LDS -> A-fragment ----
#pragma unroll
    for (int r = 0; r < 8; r++) {
      Pbuf[wave][r + hi * 8][col] = (bf16_t)p0[r];
      Pbuf[wave][r + hi * 8][16 + col] = (bf16_t)p1[r];
    }
    asm volatile("s_wait_dscnt 0" ::: "memory");  // per-wave LDS RAW fence
    v16bf pa = wmma_a_rowmajor(&Pbuf[wave][0][0], JTILE, lane);

    // ---- O += P * V^T over 16 channel tiles (K=32, B from LDS) ----
#pragma unroll
    for (int t = 0; t < CCH / 16; t++) {
      v16bf vb = wmma_b_kcontig(&Vtile[p][t * 16][0], JTILE, lane);
      acc[t] = wmma_bf16(pa, vb, acc[t]);
    }

    __syncthreads();  // all waves done with buffer p before re-staging it
    if (jt + 2 < NT) issue_tile(jt + 2, p);
  }

  // ---- epilogue: out = gamma * (O / l) + inFeature ----
  const float g = gamma[0];
  float rl[8];
#pragma unroll
  for (int r = 0; r < 8; r++) rl[r] = 1.f / lrow[r];
#pragma unroll
  for (int t = 0; t < CCH / 16; t++) {
    int c = t * 16 + col;
    size_t base = ((size_t)b * CCH + c) * NSP + r0 + hi * 8;
#pragma unroll
    for (int r = 0; r < 8; r++) {
      float o = acc[t][r] * rl[r];
      out[base + r] = g * o + inF[base + r];
    }
  }
}

// ---------------------------------------------------------------------------
extern "C" void kernel_launch(void* const* d_in, const int* in_sizes, int n_in,
                              void* d_out, int out_size, void* d_ws, size_t ws_size,
                              hipStream_t stream) {
  (void)in_sizes; (void)n_in; (void)out_size; (void)ws_size;
  const float* inF = (const float*)d_in[0];
  const float* Wq = (const float*)d_in[1];
  const float* bq = (const float*)d_in[2];
  const float* Wk = (const float*)d_in[3];
  const float* bk = (const float*)d_in[4];
  const float* Wv = (const float*)d_in[5];
  const float* bv = (const float*)d_in[6];
  const float* gamma = (const float*)d_in[7];
  float* out = (float*)d_out;

  char* ws = (char*)d_ws;
  const size_t wv_bytes = (size_t)CCH * CCH * sizeof(bf16_t);         // 128 KB
  const size_t qk_bytes = (size_t)BATCH * NSP * CQK * sizeof(bf16_t); // 2.25 MB
  bf16_t* Wv_bf = (bf16_t*)ws;
  bf16_t* Qbf = (bf16_t*)(ws + wv_bytes);
  bf16_t* Kbf = (bf16_t*)(ws + wv_bytes + qk_bytes);
  bf16_t* Vbf = (bf16_t*)(ws + wv_bytes + 2 * qk_bytes);              // 18 MB

  hipLaunchKernelGGL(cvt_kernel, dim3((CCH * CCH + 255) / 256), dim3(256), 0,
                     stream, Wv, Wv_bf, CCH * CCH);
  hipLaunchKernelGGL(qkproj_kernel, dim3(BATCH * NSP / 256), dim3(256), 0,
                     stream, inF, Wq, bq, Wk, bk, Qbf, Kbf);
  hipLaunchKernelGGL(vproj_kernel, dim3(BATCH * (NSP / 16)), dim3(256), 0,
                     stream, inF, Wv_bf, bv, Vbf);
  hipLaunchKernelGGL(attn_kernel, dim3(BATCH * RBLK / NWAVES), dim3(256), 0,
                     stream, Qbf, Kbf, Vbf, inF, gamma, out);
}
